// WindowAttention_67628555043019
// MI455X (gfx1250) — matrix operand order repaired and checked by reading, not compiled, not run on hardware
//
#include <hip/hip_runtime.h>

// MI455X / gfx1250 fused Swin window attention.
// All four GEMMs (QKV, Q@K^T, P@V, proj) run on v_wmma_f32_16x16x32_f16.

typedef __attribute__((ext_vector_type(16))) _Float16 v16h;
typedef __attribute__((ext_vector_type(8)))  float    v8f;

#define NTOK   49
#define DIM    384
#define NH     12
#define HD     32
#define NB     2048
#define QSCALE 0.17677669529663687f   // 32^-0.5

union Frag16 { v16h v; uint4 q[2]; };
union Half4  { _Float16 h[4]; unsigned long long u; };
union Half8  { _Float16 h[8]; uint4 q; };

// 32 contiguous bytes per lane (B-fragment style, or pre-packed weights)
__device__ __forceinline__ v16h ld_frag_cont(const _Float16* p) {
  Frag16 f;
  f.q[0] = *(const uint4*)(p);
  f.q[1] = *(const uint4*)(p + 8);
  return f.v;
}
// A-fragment: halves 0..7 at p, halves 8..15 at p+16 (K split lo/hi 16)
__device__ __forceinline__ v16h ld_frag_a(const _Float16* p) {
  Frag16 f;
  f.q[0] = *(const uint4*)(p);
  f.q[1] = *(const uint4*)(p + 16);
  return f.v;
}
__device__ __forceinline__ v8f zero8() {
  v8f z = {0.f, 0.f, 0.f, 0.f, 0.f, 0.f, 0.f, 0.f};
  return z;
}
__device__ __forceinline__ v8f wmma_f16(v16h a, v16h b, v8f c) {
  return __builtin_amdgcn_wmma_f32_16x16x32_f16(false, a, false, b, (short)0, c,
                                                false, false);
}

// ---------------------------------------------------------------------------
// Pack fp32 weights into f16 fragment-linear layout:
//   w[(kk*NT + nt)*512 + lane*16 + i] = W[kk*32 + (i<8 ? 8*(lane>>4)+i : 16+8*(lane>>4)+i-8)][nt*16 + (lane&15)]
// so a B-fragment is one 32B contiguous read per lane.
// ---------------------------------------------------------------------------
__global__ void pack_w(const float* __restrict__ qkv_w,
                       const float* __restrict__ proj_w,
                       _Float16* __restrict__ wq,
                       _Float16* __restrict__ wp) {
  const int NQ = 12 * 72 * 512;   // K=384 (12 ksteps) x N=1152 (72 tiles)
  const int NP = 12 * 24 * 512;   // K=384 x N=384 (24 tiles)
  for (int t = blockIdx.x * blockDim.x + threadIdx.x; t < NQ + NP;
       t += gridDim.x * blockDim.x) {
    if (t < NQ) {
      int i = t & 15, lane = (t >> 4) & 31, tile = t >> 9;
      int nt = tile % 72, kk = tile / 72;
      int k = kk * 32 + ((i < 8) ? ((lane >> 4) << 3) + i : 16 + ((lane >> 4) << 3) + i - 8);
      int n = nt * 16 + (lane & 15);
      wq[t] = (_Float16)qkv_w[k * 1152 + n];
    } else {
      int u = t - NQ;
      int i = u & 15, lane = (u >> 4) & 31, tile = u >> 9;
      int nt = tile % 24, kk = tile / 24;
      int k = kk * 32 + ((i < 8) ? ((lane >> 4) << 3) + i : 16 + ((lane >> 4) << 3) + i - 8);
      int n = nt * 16 + (lane & 15);
      wp[u] = (_Float16)proj_w[k * 384 + n];
    }
  }
}

// ---------------------------------------------------------------------------
// Fused per-window kernel. One block per window, 256 threads = 8 waves.
// LDS map (halves):
//   [0 .. 36864)            : Xs [64][392] (phases 1-2), then Ps 8x[64][72] (phase 3)
//   [36864 .. 61952)        : Qs [64][392], overwritten per-head with O
//   [61952 .. 87040)        : Ks [64][392]
//   [87040 .. 114688)       : Vt [384][72]  (V transposed: [head-dim][token])
// ---------------------------------------------------------------------------
__global__ __launch_bounds__(256) void win_attn(
    const float* __restrict__ x, const float* __restrict__ attn_mask,
    const float* __restrict__ qkv_b, const float* __restrict__ rpb,
    const float* __restrict__ proj_b, const _Float16* __restrict__ wq,
    const _Float16* __restrict__ wp, float* __restrict__ out) {
  __shared__ __align__(16) _Float16 smem[114688];
  _Float16* Xs     = smem;
  _Float16* Ps_all = smem;
  _Float16* Qs     = smem + 36864;
  _Float16* Ks     = smem + 61952;
  _Float16* Vt     = smem + 87040;

  const int tid  = threadIdx.x;
  const int lane = tid & 31;
  const int wave = tid >> 5;
  const int l15  = lane & 15;
  const int hi   = lane >> 4;    // which 16-lane half
  const int hi8  = hi << 3;
  const int b    = blockIdx.x;

  // ---- Phase 1: load x window -> f16 LDS (vectorized), zero rows 49..63 ----
  for (int t = tid; t < 64 * (DIM / 4); t += 256) {
    int r = t / 96, c4 = (t - r * 96) * 4;
    float4 v = make_float4(0.f, 0.f, 0.f, 0.f);
    if (r < NTOK)
      v = *(const float4*)(x + ((size_t)b * NTOK + r) * DIM + c4);
    Half4 p;
    p.h[0] = (_Float16)v.x;
    p.h[1] = (_Float16)v.y;
    p.h[2] = (_Float16)v.z;
    p.h[3] = (_Float16)v.w;
    *(unsigned long long*)(Xs + r * 392 + c4) = p.u;
  }
  __syncthreads();

  // ---- Phase 2: QKV = X @ Wqkv (+bias). M=64, N=1152, K=384 ----
  for (int nt = wave; nt < 72; nt += 8) {
    v8f acc0 = zero8(), acc1 = zero8(), acc2 = zero8(), acc3 = zero8();
    for (int kk = 0; kk < 12; ++kk) {
      v16h bf = ld_frag_cont(wq + (size_t)(kk * 72 + nt) * 512 + lane * 16);
      const _Float16* ab = Xs + l15 * 392 + kk * 32 + hi8;
      v16h a0 = ld_frag_a(ab + 0 * 16 * 392);
      v16h a1 = ld_frag_a(ab + 1 * 16 * 392);
      v16h a2 = ld_frag_a(ab + 2 * 16 * 392);
      v16h a3 = ld_frag_a(ab + 3 * 16 * 392);
      acc0 = wmma_f16(a0, bf, acc0);
      acc1 = wmma_f16(a1, bf, acc1);
      acc2 = wmma_f16(a2, bf, acc2);
      acc3 = wmma_f16(a3, bf, acc3);
    }
    const int ncol = nt * 16 + l15;
    const float bb = qkv_b[ncol];
    if (nt < 24) {               // Q (scaled), per-lane column scatter
#pragma unroll
      for (int v = 0; v < 8; ++v) {
        Qs[( 0 + v + hi8) * 392 + ncol] = (_Float16)((acc0[v] + bb) * QSCALE);
        Qs[(16 + v + hi8) * 392 + ncol] = (_Float16)((acc1[v] + bb) * QSCALE);
        Qs[(32 + v + hi8) * 392 + ncol] = (_Float16)((acc2[v] + bb) * QSCALE);
        Qs[(48 + v + hi8) * 392 + ncol] = (_Float16)((acc3[v] + bb) * QSCALE);
      }
    } else if (nt < 48) {        // K
      const int ck = ncol - 384;
#pragma unroll
      for (int v = 0; v < 8; ++v) {
        Ks[( 0 + v + hi8) * 392 + ck] = (_Float16)(acc0[v] + bb);
        Ks[(16 + v + hi8) * 392 + ck] = (_Float16)(acc1[v] + bb);
        Ks[(32 + v + hi8) * 392 + ck] = (_Float16)(acc2[v] + bb);
        Ks[(48 + v + hi8) * 392 + ck] = (_Float16)(acc3[v] + bb);
      }
    } else {                     // V -> Vt[head-dim][token]; rows contiguous in
      const int cv = ncol - 768; //  token dim: pack 8 halves -> one b128 store
      Half8 p0, p1, p2, p3;
#pragma unroll
      for (int v = 0; v < 8; ++v) {
        p0.h[v] = (_Float16)(acc0[v] + bb);
        p1.h[v] = (_Float16)(acc1[v] + bb);
        p2.h[v] = (_Float16)(acc2[v] + bb);
        p3.h[v] = (_Float16)(acc3[v] + bb);
      }
      *(uint4*)(Vt + cv * 72 +  0 + hi8) = p0.q;
      *(uint4*)(Vt + cv * 72 + 16 + hi8) = p1.q;
      *(uint4*)(Vt + cv * 72 + 32 + hi8) = p2.q;
      *(uint4*)(Vt + cv * 72 + 48 + hi8) = p3.q;
    }
  }
  __syncthreads();

  // ---- Phase 3: per-head attention. Heads round-robined over waves ----
  const int wwin = b & 63;
  const float* maskp = attn_mask + (size_t)wwin * NTOK * NTOK;
  _Float16* Ps = Ps_all + wave * 4608;   // [64][72] per-wave P tile

  for (int h = wave; h < NH; h += 8) {
    for (int it = 0; it < 4; ++it) {
      // S row-block: 4 WMMAs, K = head-dim = 32 exactly
      v16h qa = ld_frag_a(Qs + (it * 16 + l15) * 392 + h * 32 + hi8);
      v16h kb0 = ld_frag_a(Ks + ( 0 + l15) * 392 + h * 32 + hi8);
      v16h kb1 = ld_frag_a(Ks + (16 + l15) * 392 + h * 32 + hi8);
      v16h kb2 = ld_frag_a(Ks + (32 + l15) * 392 + h * 32 + hi8);
      v16h kb3 = ld_frag_a(Ks + (48 + l15) * 392 + h * 32 + hi8);
      v8f s0 = wmma_f16(qa, kb0, zero8());
      v8f s1 = wmma_f16(qa, kb1, zero8());
      v8f s2 = wmma_f16(qa, kb2, zero8());
      v8f s3 = wmma_f16(qa, kb3, zero8());

      // bias + mask + masked softmax (row lives in a 16-lane half)
#pragma unroll
      for (int v = 0; v < 8; ++v) {
        int r = it * 16 + v + hi8;
        float t0 = s0[v], t1 = s1[v], t2 = s2[v], t3 = s3[v];
        if (r < NTOK) {
          int ri = r / 7, rj = r - ri * 7;
          {
            int c = l15;
            if (c < NTOK) {
              int ci = c / 7, cj = c - ci * 7;
              t0 += rpb[((ri - ci + 6) * 13 + (rj - cj + 6)) * NH + h] +
                    maskp[r * NTOK + c];
            } else t0 = -1e30f;
          }
          {
            int c = 16 + l15;
            int ci = c / 7, cj = c - ci * 7;
            t1 += rpb[((ri - ci + 6) * 13 + (rj - cj + 6)) * NH + h] +
                  maskp[r * NTOK + c];
          }
          {
            int c = 32 + l15;
            int ci = c / 7, cj = c - ci * 7;
            t2 += rpb[((ri - ci + 6) * 13 + (rj - cj + 6)) * NH + h] +
                  maskp[r * NTOK + c];
          }
          {
            int c = 48 + l15;
            if (c < NTOK) {
              int ci = c / 7, cj = c - ci * 7;
              t3 += rpb[((ri - ci + 6) * 13 + (rj - cj + 6)) * NH + h] +
                    maskp[r * NTOK + c];
            } else t3 = -1e30f;
          }
        } else {
          t0 = t1 = t2 = t3 = -1e30f;
        }
        float m = fmaxf(fmaxf(t0, t1), fmaxf(t2, t3));
        m = fmaxf(m, __shfl_xor(m, 1, 16));
        m = fmaxf(m, __shfl_xor(m, 2, 16));
        m = fmaxf(m, __shfl_xor(m, 4, 16));
        m = fmaxf(m, __shfl_xor(m, 8, 16));
        float e0 = __expf(t0 - m), e1 = __expf(t1 - m);
        float e2 = __expf(t2 - m), e3 = __expf(t3 - m);
        float sum = e0 + e1 + e2 + e3;
        sum += __shfl_xor(sum, 1, 16);
        sum += __shfl_xor(sum, 2, 16);
        sum += __shfl_xor(sum, 4, 16);
        sum += __shfl_xor(sum, 8, 16);
        float inv = 1.0f / sum;
        _Float16* pr = Ps + r * 72;
        pr[ 0 + l15] = (_Float16)(e0 * inv);
        pr[16 + l15] = (_Float16)(e1 * inv);
        pr[32 + l15] = (_Float16)(e2 * inv);
        pr[48 + l15] = (_Float16)(e3 * inv);
      }

      // O row-block = P @ V_h : M=16, N=32, K=64 (2 ksteps)
      v8f o0 = zero8(), o1 = zero8();
#pragma unroll
      for (int ks = 0; ks < 2; ++ks) {
        v16h pa  = ld_frag_a(Ps + (it * 16 + l15) * 72 + ks * 32 + hi8);
        v16h vb0 = ld_frag_a(Vt + (h * 32 +  0 + l15) * 72 + ks * 32 + hi8);
        v16h vb1 = ld_frag_a(Vt + (h * 32 + 16 + l15) * 72 + ks * 32 + hi8);
        o0 = wmma_f16(pa, vb0, o0);
        o1 = wmma_f16(pa, vb1, o1);
      }
      // O overwrites consumed Q columns of this head (same rows only)
#pragma unroll
      for (int v = 0; v < 8; ++v) {
        int r = it * 16 + v + hi8;
        Qs[r * 392 + h * 32 + l15]      = (_Float16)o0[v];
        Qs[r * 392 + h * 32 + 16 + l15] = (_Float16)o1[v];
      }
    }
  }
  __syncthreads();

  // ---- Phase 4: out = O @ Wproj + b. M=64, N=384, K=384 ----
  for (int nt = wave; nt < 24; nt += 8) {
    v8f acc0 = zero8(), acc1 = zero8(), acc2 = zero8(), acc3 = zero8();
    for (int kk = 0; kk < 12; ++kk) {
      v16h bf = ld_frag_cont(wp + (size_t)(kk * 24 + nt) * 512 + lane * 16);
      const _Float16* ab = Qs + l15 * 392 + kk * 32 + hi8;
      v16h a0 = ld_frag_a(ab + 0 * 16 * 392);
      v16h a1 = ld_frag_a(ab + 1 * 16 * 392);
      v16h a2 = ld_frag_a(ab + 2 * 16 * 392);
      v16h a3 = ld_frag_a(ab + 3 * 16 * 392);
      acc0 = wmma_f16(a0, bf, acc0);
      acc1 = wmma_f16(a1, bf, acc1);
      acc2 = wmma_f16(a2, bf, acc2);
      acc3 = wmma_f16(a3, bf, acc3);
    }
    const int col = nt * 16 + l15;
    const float pb = proj_b[col];
#pragma unroll
    for (int v = 0; v < 8; ++v) {
      int r0 = v + hi8;
      if (r0 < NTOK) out[((size_t)b * NTOK + r0) * DIM + col] = acc0[v] + pb;
      int r1 = 16 + r0;
      if (r1 < NTOK) out[((size_t)b * NTOK + r1) * DIM + col] = acc1[v] + pb;
      int r2 = 32 + r0;
      if (r2 < NTOK) out[((size_t)b * NTOK + r2) * DIM + col] = acc2[v] + pb;
      int r3 = 48 + r0;
      if (r3 < NTOK) out[((size_t)b * NTOK + r3) * DIM + col] = acc3[v] + pb;
    }
  }
}

extern "C" void kernel_launch(void* const* d_in, const int* in_sizes, int n_in,
                              void* d_out, int out_size, void* d_ws,
                              size_t ws_size, hipStream_t stream) {
  const float* x      = (const float*)d_in[0];
  const float* mask   = (const float*)d_in[1];
  const float* qkv_w  = (const float*)d_in[2];
  const float* qkv_b  = (const float*)d_in[3];
  const float* rpb    = (const float*)d_in[4];
  const float* proj_w = (const float*)d_in[5];
  const float* proj_b = (const float*)d_in[6];
  float* out = (float*)d_out;

  _Float16* wq = (_Float16*)d_ws;        // 442368 halves = 884736 B
  _Float16* wp = wq + 442368;            // 147456 halves = 294912 B

  pack_w<<<1152, 256, 0, stream>>>(qkv_w, proj_w, wq, wp);
  win_attn<<<NB, 256, 0, stream>>>(x, mask, qkv_b, rpb, proj_b, wq, wp, out);
}
